// PoseGNN_69630009803228
// MI455X (gfx1250) — compile-verified, matrix operand order checked by reading, not run on hardware
//
#include <hip/hip_runtime.h>
#include <hip/hip_bf16.h>

// ---------------------------------------------------------------------------
// PoseGNN forward on gfx1250 (MI455X), round 5.
//  - Memory-bound pipeline: 410 MB fp32 `x` stream dominates (~18 us @ 23.3 TB/s);
//    h/agg buffers (<= 25.6 MB) and edge lists (12.8 MB) are L2-resident.
//  - fp32 operands split on the fly into bf16 hi/lo, 3x v_wmma_f32_16x16x32_bf16
//    per K-step (hi*hi + lo*hi + hi*lo) ~= fp32 accuracy on the fast bf16 pipe.
//  - One wave computes a 16 x (16*NT) output strip: A fragment loaded+split once
//    per K-step, reused for all NT N-tiles.
//  - Weights packed s-major fragment-major: b128 loads with immediate offsets,
//    clause-grouped; W fragments register-buffered in chunks sized to keep the
//    NT=8 kernel under the VGPR budget (no acc spills: chunk=2 when NT=8).
// ---------------------------------------------------------------------------

typedef __attribute__((ext_vector_type(16))) __bf16 bf16x16;
typedef __attribute__((ext_vector_type(8)))  float  f32x8;
typedef __attribute__((ext_vector_type(4)))  float  f32x4;

#define N_NODES 50000
#define N_EDGES 1600000
#define LRELU_SLOPE 0.01f

// 16-bit A/B fragment k-offsets (ISA 7.12.2): v16bf element j of lane L holds
// K = kk0 + {0..7,16..23}[j], kk0 = 32*s + 8*(L>=16); row/col = L & 15.
__device__ __forceinline__ int frag_koff_pair(int i) {   // i = element pair 0..7
    return (i & 3) * 2 + (i >> 2) * 16;
}

// Load 16 fp32 A values (two contiguous 8-float runs -> 4x global_load_b128)
// and split into bf16 hi/lo fragments.  `ap` already includes 8*half.
template <bool NTA>
__device__ __forceinline__ void load_a_split(const float* __restrict__ ap,
                                             bf16x16& hi, bf16x16& lo) {
    const f32x4* p0 = reinterpret_cast<const f32x4*>(ap);
    const f32x4* p1 = reinterpret_cast<const f32x4*>(ap + 16);
    f32x4 v0, v1, v2, v3;
    if (NTA) {                                  // streamed input: TH=NT
        v0 = __builtin_nontemporal_load(p0);
        v1 = __builtin_nontemporal_load(p0 + 1);
        v2 = __builtin_nontemporal_load(p1);
        v3 = __builtin_nontemporal_load(p1 + 1);
    } else {
        v0 = p0[0]; v1 = p0[1]; v2 = p1[0]; v3 = p1[1];
    }
    float f[16] = {v0.x, v0.y, v0.z, v0.w, v1.x, v1.y, v1.z, v1.w,
                   v2.x, v2.y, v2.z, v2.w, v3.x, v3.y, v3.z, v3.w};
#pragma unroll
    for (int j = 0; j < 16; ++j) {
        __bf16 h = (__bf16)f[j];
        hi[j] = h;
        lo[j] = (__bf16)(f[j] - (float)h);
    }
}

// Accumulate A[16-row strip] @ W^T into NT 16x16 tiles.
// Packed weight layout (s-major): fragment (s,t) of lane L at ((s*NT+t)*32+L)*16.
template <int NT, bool NTA>
__device__ __forceinline__ void gemm_strip(const float* __restrict__ A,
                                           const __bf16* __restrict__ WhiP,
                                           const __bf16* __restrict__ WloP,
                                           int K, int row, int lane, int half,
                                           f32x8* acc) {
    const int S = K >> 5;
    const float* ap = A + (size_t)row * K + 8 * half;
    const bf16x16* WH = reinterpret_cast<const bf16x16*>(WhiP) + lane;
    const bf16x16* WL = reinterpret_cast<const bf16x16*>(WloP) + lane;
    // W register-buffer chunk: keep NT=8 under the VGPR budget (acc alone = 64).
    constexpr int C = (NT >= 8) ? 2 : ((NT < 4) ? NT : 4);
    for (int s = 0; s < S; ++s) {
        if (s + 1 < S) __builtin_prefetch(ap + 32, 0, 3);   // global_prefetch_b8
        bf16x16 ahi, alo;
        load_a_split<NTA>(ap, ahi, alo);
        ap += 32;
#pragma unroll
        for (int tc = 0; tc < NT; tc += C) {
            bf16x16 wh[C], wl[C];
#pragma unroll
            for (int u = 0; u < C; ++u) {       // clause of b128 loads, imm offsets
                wh[u] = WH[(size_t)(tc + u) * 32];
                wl[u] = WL[(size_t)(tc + u) * 32];
            }
#pragma unroll
            for (int u = 0; u < C; ++u) {       // 3*C WMMAs back-to-back
                acc[tc + u] = __builtin_amdgcn_wmma_f32_16x16x32_bf16(
                    false, ahi, false, wh[u], (short)0, acc[tc + u], false, false);
                acc[tc + u] = __builtin_amdgcn_wmma_f32_16x16x32_bf16(
                    false, alo, false, wh[u], (short)0, acc[tc + u], false, false);
                acc[tc + u] = __builtin_amdgcn_wmma_f32_16x16x32_bf16(
                    false, ahi, false, wl[u], (short)0, acc[tc + u], false, false);
            }
        }
        WH += (size_t)NT * 32;                  // next K-step block
        WL += (size_t)NT * 32;
    }
}

// Fused strip GEMM: OUT[M,Nreal] = act(A1@W1^T (+ A2@W2^T) + bias).
// One wave per 16-row strip; NT = Npad/16 accumulator tiles per wave.
template <int NT, bool NTA>
__global__ void __launch_bounds__(256)
wmma_gemm_kernel(const float* __restrict__ A1,
                 const __bf16* __restrict__ W1hi,
                 const __bf16* __restrict__ W1lo, int K1,
                 const float* __restrict__ A2,
                 const __bf16* __restrict__ W2hi,
                 const __bf16* __restrict__ W2lo, int K2,
                 const float* __restrict__ bias,
                 float* __restrict__ OUT,
                 int M, int Nreal, int applyLrelu) {
    const int lane  = threadIdx.x & 31;
    const int wave  = threadIdx.x >> 5;
    const int tileM = blockIdx.x * (blockDim.x >> 5) + wave;
    if (tileM >= (M >> 4)) return;          // wave-uniform: EXEC stays all-1s
    const int half = lane >> 4;
    const int l16  = lane & 15;
    const int row  = tileM * 16 + l16;      // A fragment row (M = lane&15)

    f32x8 acc[NT];
#pragma unroll
    for (int t = 0; t < NT; ++t) {
        f32x8 z = {0.f, 0.f, 0.f, 0.f, 0.f, 0.f, 0.f, 0.f};
        acc[t] = z;
    }

    gemm_strip<NT, NTA>(A1, W1hi, W1lo, K1, row, lane, half, acc);
    if (A2) gemm_strip<NT, false>(A2, W2hi, W2lo, K2, row, lane, half, acc);

#pragma unroll
    for (int t = 0; t < NT; ++t) {
        const int col = t * 16 + l16;
        if (col < Nreal) {                  // only trims the padded pos layer
            const float b = bias ? bias[col] : 0.0f;
#pragma unroll
            for (int r = 0; r < 8; ++r) {   // C/D layout: M = r + 8*half
                int orow = tileM * 16 + r + 8 * half;
                float v = acc[t][r] + b;
                if (applyLrelu) v = (v >= 0.0f) ? v : LRELU_SLOPE * v;
                OUT[(size_t)orow * Nreal + col] = v;
            }
        }
    }
}

// Pre-swizzle fp32 weights [Nrows,K] into s-major fragment-major bf16 hi/lo
// planes of Npad rows (zero-padded).  idx = ((s*T + t)*32 + lane)*16 + j.
__global__ void pack_wfrag_kernel(const float* __restrict__ W,
                                  __bf16* __restrict__ hi, __bf16* __restrict__ lo,
                                  int Nrows, int Npad, int K) {
    int idx = blockIdx.x * blockDim.x + threadIdx.x;
    if (idx >= Npad * K) return;
    const int T    = Npad >> 4;
    const int j    = idx & 15;
    const int lane = (idx >> 4) & 31;
    const int st   = idx >> 9;
    const int t    = st % T;
    const int s    = st / T;
    const int i    = j >> 1;
    const int row  = t * 16 + (lane & 15);
    const int col  = (s << 5) + ((lane >> 4) << 3) + frag_koff_pair(i) + (j & 1);
    float f = (row < Nrows) ? W[(size_t)row * K + col] : 0.0f;
    __bf16 h = (__bf16)f;
    hi[idx] = h;
    lo[idx] = (__bf16)(f - (float)h);
}

// Segment-sum: agg[dst[e], f..f+3] += h[src[e], f..f+3].  F/4 lanes per edge
// (wave-uniform edge id for F=128) -> coalesced b128 gathers + f32 atomics
// that resolve in the L2-resident agg buffer.
__global__ void __launch_bounds__(256)
scatter_add_kernel(const float* __restrict__ h,
                   const int* __restrict__ src,
                   const int* __restrict__ dst,
                   float* __restrict__ agg,
                   int nEdges, int logQ) {   // logQ = log2(F/4)
    size_t idx = (size_t)blockIdx.x * blockDim.x + threadIdx.x;
    int e = (int)(idx >> logQ);
    if (e >= nEdges) return;
    int q = (int)(idx & ((1u << logQ) - 1u));
    int F = 4 << logQ;
    int s = src[e];
    int d = dst[e];
    f32x4 v = *reinterpret_cast<const f32x4*>(h + (size_t)s * F + q * 4);
    float* out = agg + (size_t)d * F + q * 4;
    atomicAdd(out + 0, v.x);
    atomicAdd(out + 1, v.y);
    atomicAdd(out + 2, v.z);
    atomicAdd(out + 3, v.w);
}

static inline int strip_blocks(int M) {
    int strips = M / 16;                    // 3125, exact
    return (strips + 7) / 8;                // 8 waves per 256-thread block
}

extern "C" void kernel_launch(void* const* d_in, const int* in_sizes, int n_in,
                              void* d_out, int out_size, void* d_ws, size_t ws_size,
                              hipStream_t stream) {
    const float* x        = (const float*)d_in[0];
    const int*   ei       = (const int*)d_in[1];     // [2, N_EDGES]
    const float* lin1_w   = (const float*)d_in[2];   // [128, 2048]
    const float* lin1_b   = (const float*)d_in[3];
    const float* c1_wrel  = (const float*)d_in[4];   // [64, 128]
    const float* c1_brel  = (const float*)d_in[5];
    const float* c1_wroot = (const float*)d_in[6];   // [64, 128]
    const float* c4_wrel  = (const float*)d_in[7];   // [32, 64]
    const float* c4_brel  = (const float*)d_in[8];
    const float* c4_wroot = (const float*)d_in[9];   // [32, 64]
    const float* pos_w    = (const float*)d_in[10];  // [3, 32] -> padded [16,32]
    const float* pos_b    = (const float*)d_in[11];
    (void)in_sizes; (void)n_in; (void)out_size; (void)ws_size;

    const int* e_src = ei;
    const int* e_dst = ei + N_EDGES;

    // ---- workspace layout (all chunks are multiples of 256 B) ----
    char* ws = (char*)d_ws;
    size_t off = 0;
    auto alloc = [&](size_t bytes) { char* p = ws + off; off += bytes; return p; };

    float* h1   = (float*)alloc((size_t)N_NODES * 128 * 4);
    float* agg1 = (float*)alloc((size_t)N_NODES * 128 * 4);
    float* h2   = (float*)alloc((size_t)N_NODES * 64 * 4);
    float* agg2 = (float*)alloc((size_t)N_NODES * 64 * 4);
    float* h3   = (float*)alloc((size_t)N_NODES * 32 * 4);

    __bf16* w1_hi = (__bf16*)alloc((size_t)128 * 2048 * 2);
    __bf16* w1_lo = (__bf16*)alloc((size_t)128 * 2048 * 2);
    __bf16* r1_hi = (__bf16*)alloc((size_t)64 * 128 * 2);
    __bf16* r1_lo = (__bf16*)alloc((size_t)64 * 128 * 2);
    __bf16* g1_hi = (__bf16*)alloc((size_t)64 * 128 * 2);
    __bf16* g1_lo = (__bf16*)alloc((size_t)64 * 128 * 2);
    __bf16* r4_hi = (__bf16*)alloc((size_t)32 * 64 * 2);
    __bf16* r4_lo = (__bf16*)alloc((size_t)32 * 64 * 2);
    __bf16* g4_hi = (__bf16*)alloc((size_t)32 * 64 * 2);
    __bf16* g4_lo = (__bf16*)alloc((size_t)32 * 64 * 2);
    __bf16* p_hi  = (__bf16*)alloc((size_t)16 * 32 * 2);
    __bf16* p_lo  = (__bf16*)alloc((size_t)16 * 32 * 2);

    // ---- 0) weight prep: split + swizzle into fragment-major bf16 hi/lo ----
    auto pack = [&](const float* w, __bf16* hi, __bf16* lo, int Nrows, int Npad, int K) {
        int n = Npad * K;
        pack_wfrag_kernel<<<(n + 255) / 256, 256, 0, stream>>>(w, hi, lo, Nrows, Npad, K);
    };
    pack(lin1_w,   w1_hi, w1_lo, 128, 128, 2048);
    pack(c1_wrel,  r1_hi, r1_lo, 64, 64, 128);
    pack(c1_wroot, g1_hi, g1_lo, 64, 64, 128);
    pack(c4_wrel,  r4_hi, r4_lo, 32, 32, 64);
    pack(c4_wroot, g4_hi, g4_lo, 32, 32, 64);
    pack(pos_w,    p_hi,  p_lo,  3, 16, 32);         // zero-padded rows 3..15

    const int gb = strip_blocks(N_NODES);            // 391 blocks of 8 waves

    // ---- 1) h1 = lrelu(x @ lin1_w^T + b1)  [50000,128], K=2048 (NT streamed) ----
    wmma_gemm_kernel<8, true><<<gb, 256, 0, stream>>>(
        x, w1_hi, w1_lo, 2048,
        nullptr, nullptr, nullptr, 0,
        lin1_b, h1, N_NODES, 128, 1);

    // ---- 2) agg1 = segment_sum(h1[src] -> dst), F=128 ----
    (void)hipMemsetAsync(agg1, 0, (size_t)N_NODES * 128 * 4, stream);
    {
        size_t total = (size_t)N_EDGES << 5;         // F/4 = 32 lanes per edge
        scatter_add_kernel<<<(unsigned)((total + 255) / 256), 256, 0, stream>>>(
            h1, e_src, e_dst, agg1, N_EDGES, 5);
    }

    // ---- 3) h2 = lrelu(agg1 @ Wrel^T + h1 @ Wroot^T + b)  [50000,64] ----
    wmma_gemm_kernel<4, false><<<gb, 256, 0, stream>>>(
        agg1, r1_hi, r1_lo, 128,
        h1,   g1_hi, g1_lo, 128,
        c1_brel, h2, N_NODES, 64, 1);

    // ---- 4) agg2 = segment_sum(h2[src] -> dst), F=64 ----
    (void)hipMemsetAsync(agg2, 0, (size_t)N_NODES * 64 * 4, stream);
    {
        size_t total = (size_t)N_EDGES << 4;         // F/4 = 16 lanes per edge
        scatter_add_kernel<<<(unsigned)((total + 255) / 256), 256, 0, stream>>>(
            h2, e_src, e_dst, agg2, N_EDGES, 4);
    }

    // ---- 5) h3 = lrelu(agg2 @ Wrel^T + h2 @ Wroot^T + b)  [50000,32] ----
    wmma_gemm_kernel<2, false><<<gb, 256, 0, stream>>>(
        agg2, r4_hi, r4_lo, 64,
        h2,   g4_hi, g4_lo, 64,
        c4_brel, h3, N_NODES, 32, 1);

    // ---- 6) position = h3 @ pos_w^T + pos_b  [50000,3] (N padded to 16) ----
    wmma_gemm_kernel<1, false><<<gb, 256, 0, stream>>>(
        h3, p_hi, p_lo, 32,
        nullptr, nullptr, nullptr, 0,
        pos_b, (float*)d_out, N_NODES, 3, 0);
}
